// VQVAE_34153579937812
// MI455X (gfx1250) — matrix operand order repaired and checked by reading, not compiled, not run on hardware
//
#include <hip/hip_runtime.h>

typedef __attribute__((ext_vector_type(16))) _Float16 v16h;
typedef __attribute__((ext_vector_type(8)))  _Float16 v8h;
typedef __attribute__((ext_vector_type(8)))  float    v8f;

#define DEVI static __device__ __forceinline__

DEVI v8f wmma32(v16h a, v16h b, v8f c) {
  // D = A(16x32 f16) * B(32x16 f16) + C(16x16 f32)
  return __builtin_amdgcn_wmma_f32_16x16x32_f16(false, a, false, b, (short)0, c, false, false);
}

DEVI v16h cat8(v8h lo, v8h hi) {
  return __builtin_shufflevector(lo, hi, 0,1,2,3,4,5,6,7,8,9,10,11,12,13,14,15);
}

// A-fragment (16x32 f16) element -> K mapping: contiguous k = 16*half + e
DEVI int a_k(int e, int half) { return e + 16 * half; }

// ---------------------------------------------------------------------------
// K0a: repack weights into WMMA A-fragment layout (+ f16 codebook & norms)
//   wA2r/wAd1r layout: [octile(8)][kstep(36)][lane(32)][elem(16)]
// ---------------------------------------------------------------------------
__global__ __launch_bounds__(256) void k_repack(
    const float* __restrict__ enc_w1, const float* __restrict__ enc_w2,
    const float* __restrict__ proj_w, const float* __restrict__ dproj_w,
    const float* __restrict__ dconv1_w, const float* __restrict__ codebook,
    _Float16* __restrict__ wA1, _Float16* __restrict__ wA2r,
    _Float16* __restrict__ wAd1r, _Float16* __restrict__ wAp,
    _Float16* __restrict__ wAdp, _Float16* __restrict__ cbf16,
    float* __restrict__ cnorm)
{
  int T = blockIdx.x * 256 + threadIdx.x;
  if (T < 9216) {                       // wA2r
    int lane = T & 31, t = (T >> 5) & 7, s = T >> 8;
    int half = lane >> 4, oc = t * 16 + (lane & 15);
    size_t dst = ((size_t)(t * 36 + s) * 32 + lane) * 16;
    for (int e = 0; e < 16; ++e) {
      int gk = s * 32 + a_k(e, half);   // K order: tap*128 + c
      int tap = gk >> 7, c = gk & 127;
      wA2r[dst + e] = (_Float16)enc_w2[(oc * 128 + c) * 9 + tap];
    }
  } else if (T < 18432) {               // wAd1r
    int u = T - 9216;
    int lane = u & 31, t = (u >> 5) & 7, s = u >> 8;
    int half = lane >> 4, oc = t * 16 + (lane & 15);
    size_t dst = ((size_t)(t * 36 + s) * 32 + lane) * 16;
    for (int e = 0; e < 16; ++e) {
      int gk = s * 32 + a_k(e, half);
      int tap = gk >> 7, c = gk & 127;
      wAd1r[dst + e] = (_Float16)dconv1_w[(oc * 128 + c) * 9 + tap];
    }
  } else if (T < 18688) {               // wA1: [8][32][16], K=27 (tap*3+c), zero pad
    int u = T - 18432;
    int lane = u & 31, t = u >> 5;
    int half = lane >> 4, oc = t * 16 + (lane & 15);
    for (int e = 0; e < 16; ++e) {
      int gk = a_k(e, half);
      float v = 0.f;
      if (gk < 27) { int tap = gk / 3, c = gk - tap * 3; v = enc_w1[(oc * 3 + c) * 9 + tap]; }
      wA1[u * 16 + e] = (_Float16)v;
    }
  } else if (T < 19200) {               // wAp (proj 64x128): [4 ksteps][4 dtiles][32][16]
    int u = T - 18688;
    int lane = u & 31, t = (u >> 5) & 3, s = u >> 7;
    int half = lane >> 4, dd = t * 16 + (lane & 15);
    for (int e = 0; e < 16; ++e) {
      int gk = s * 32 + a_k(e, half);
      wAp[u * 16 + e] = (_Float16)proj_w[dd * 128 + gk];
    }
  } else if (T < 19712) {               // wAdp (dproj 128x64): [2 ksteps][8 htiles][32][16]
    int u = T - 19200;
    int lane = u & 31, t = (u >> 5) & 7, s = u >> 8;
    int half = lane >> 4, hc = t * 16 + (lane & 15);
    for (int e = 0; e < 16; ++e) {
      int gk = s * 32 + a_k(e, half);
      wAdp[u * 16 + e] = (_Float16)dproj_w[hc * 64 + gk];
    }
  } else if (T < 20224) {               // codebook -> f16 + squared norms
    int code = T - 19712;
    float acc = 0.f;
    for (int d = 0; d < 64; ++d) {
      float v = codebook[code * 64 + d];
      cbf16[code * 64 + d] = (_Float16)v;
      acc += v * v;
    }
    cnorm[code] = acc;
  }
}

// ---------------------------------------------------------------------------
// K0b: zero a buffer in 16B chunks (used for padded u-buffer)
// ---------------------------------------------------------------------------
__global__ __launch_bounds__(256) void k_zero(v8h* __restrict__ p, int nchunks)
{
  int t = blockIdx.x * 256 + threadIdx.x;
  if (t < nchunks) { v8h z = {}; p[t] = z; }
}

// ---------------------------------------------------------------------------
// K0c: zero the 1-pixel border of padded h1 (32,66,66,128) f16
// ---------------------------------------------------------------------------
__global__ __launch_bounds__(256) void k_border(_Float16* __restrict__ h1p)
{
  int t = blockIdx.x * 256 + threadIdx.x;   // 32*260*16 = 133120 threads
  if (t >= 133120) return;
  int chunk = t & 15;
  int q = t >> 4;
  int b = q / 260;
  int p = q - b * 260;
  int row, col;
  if (p < 66)       { row = 0;  col = p; }
  else if (p < 132) { row = 65; col = p - 66; }
  else { int r = p - 132; row = 1 + (r >> 1); col = (r & 1) * 65; }
  v8h z = {};
  *(v8h*)(h1p + ((size_t)(b * 66 + row) * 66 + col) * 128 + chunk * 8) = z;
}

// ---------------------------------------------------------------------------
// K1: enc conv1 3x3 s2 p1, 3->128, ReLU -> padded h1p NHWC f16 (32,66,66,128)
// One wave = 16oc x 16px tile, single WMMA (K=27 padded to 32). Branchless
// gather: clamped addresses, value-masked.
// ---------------------------------------------------------------------------
__global__ __launch_bounds__(256) void k_enc1(
    const float* __restrict__ x, const float* __restrict__ enc_b1,
    const _Float16* __restrict__ wA1, _Float16* __restrict__ h1p)
{
  int wid = blockIdx.x * 8 + (threadIdx.x >> 5);
  int lane = threadIdx.x & 31, half = lane >> 4, nl = lane & 15;
  int b = wid >> 11;                      // 2048 waves per batch
  int rem = wid & 2047;
  int pxt = rem >> 3, oct = rem & 7;

  v16h a = *(const v16h*)(wA1 + (oct * 32 + lane) * 16);

  int px = pxt * 16 + nl;                 // output pixel in 64x64
  int oy = px >> 6, ox = px & 63;
  int iy0 = 2 * oy - 1, ix0 = 2 * ox - 1;
  const float* xb = x + (size_t)b * 3 * 128 * 128;
  v8h blo, bhi;
#pragma unroll
  for (int e = 0; e < 16; ++e) {
    int k = e + 16 * half;                // B-fragment element -> K
    int tap = k / 3, c = k - tap * 3;     // c in [0,2] always
    if (tap > 8) tap = 8;
    int ki = tap / 3, kj = tap - ki * 3;
    int iy = iy0 + ki, ix = ix0 + kj;
    bool ok = (k < 27) & (iy >= 0) & (iy < 128) & (ix >= 0) & (ix < 128);
    int iyc = min(max(iy, 0), 127), ixc = min(max(ix, 0), 127);
    float v = xb[(size_t)(c * 128 + iyc) * 128 + ixc];
    _Float16 hv = (_Float16)(ok ? v : 0.f);
    if (e < 8) blo[e] = hv; else bhi[e - 8] = hv;
  }
  v16h bf = cat8(blo, bhi);
  v8f c8 = {};
  c8 = wmma32(a, bf, c8);
  int oc0 = oct * 16 + 8 * half;          // C rows: M = r + 8*half
  v8h o;
#pragma unroll
  for (int r = 0; r < 8; ++r)
    o[r] = (_Float16)fmaxf(c8[r] + enc_b1[oc0 + r], 0.f);
  *(v8h*)(h1p + ((size_t)(b * 66 + oy + 1) * 66 + ox + 1) * 128 + oc0) = o;
}

// ---------------------------------------------------------------------------
// K2: enc conv2 3x3 s2 p1, 128->128, ReLU. In: padded h1p; out: h2 NHWC f16
// (32,32,32,128). Dual pixel-tile per wave (N=32, one output row); fully
// unrolled K=1152 (9 taps x 4 channel-steps); all offsets immediate; no
// bounds checks thanks to the zero border.
// ---------------------------------------------------------------------------
__global__ __launch_bounds__(256) void k_enc2(
    const _Float16* __restrict__ h1p, const float* __restrict__ enc_b2,
    const _Float16* __restrict__ wA2r, _Float16* __restrict__ h2)
{
  int wid = blockIdx.x * 8 + (threadIdx.x >> 5);   // 8192 waves
  int lane = threadIdx.x & 31, half = lane >> 4, nl = lane & 15;
  int b = wid >> 8;                     // 256 waves per batch
  int rem = wid & 255;
  int pt = rem >> 3, oct = rem & 7;     // pt = output row (32 px)

  const _Float16* wab = wA2r + (size_t)oct * 36 * 512 + lane * 16;
  v8f c0 = {}, c1 = {};
#pragma unroll
  for (int ki = 0; ki < 3; ++ki) {
    const _Float16* lanep = h1p + ((size_t)(b * 66 + 2 * pt + ki) * 66 + 2 * nl) * 128
                                + 16 * half;
#pragma unroll
    for (int kj = 0; kj < 3; ++kj) {
      int tap = ki * 3 + kj;
#pragma unroll
      for (int cs = 0; cs < 4; ++cs) {
        v16h a  = *(const v16h*)(wab + (tap * 4 + cs) * 512);
        v16h b0 = *(const v16h*)(lanep + kj * 128 + cs * 32);
        v16h b1 = *(const v16h*)(lanep + kj * 128 + 32 * 128 + cs * 32);
        c0 = wmma32(a, b0, c0);
        c1 = wmma32(a, b1, c1);
      }
    }
  }
  int oc0 = oct * 16 + 8 * half;
  v8h o0, o1;
#pragma unroll
  for (int r = 0; r < 8; ++r) {
    float bb = enc_b2[oc0 + r];
    o0[r] = (_Float16)fmaxf(c0[r] + bb, 0.f);
    o1[r] = (_Float16)fmaxf(c1[r] + bb, 0.f);
  }
  size_t px0 = (size_t)b * 1024 + pt * 32 + nl;
  *(v8h*)(h2 + px0 * 128 + oc0) = o0;
  *(v8h*)(h2 + (px0 + 16) * 128 + oc0) = o1;
}

// ---------------------------------------------------------------------------
// K3a: adaptive avg pool 4x4: (32,32,32,128) -> pooled f16 (2048 rows, 128)
// ---------------------------------------------------------------------------
__global__ void k_pool(const _Float16* __restrict__ h2, _Float16* __restrict__ pooled)
{
  int n = blockIdx.x;                     // 0..2047 = (b, gh, gw)
  int c = threadIdx.x;                    // 0..127
  int b = n >> 6, sidx = n & 63;
  int gh = sidx >> 3, gw = sidx & 7;
  const _Float16* base = h2 + (size_t)b * 1024 * 128;
  float acc = 0.f;
  for (int i = 0; i < 4; ++i)
    for (int j = 0; j < 4; ++j)
      acc += (float)base[(size_t)((gh * 4 + i) * 32 + gw * 4 + j) * 128 + c];
  pooled[(size_t)n * 128 + c] = (_Float16)(acc * 0.0625f);
}

// ---------------------------------------------------------------------------
// K3b: proj 1x1 128->64. Writes z_e (f32 NCHW into d_out) and zf16 rows.
// ---------------------------------------------------------------------------
__global__ __launch_bounds__(256) void k_proj(
    const _Float16* __restrict__ pooled, const float* __restrict__ proj_b,
    const _Float16* __restrict__ wAp, float* __restrict__ ze_out,
    _Float16* __restrict__ zf16)
{
  int wid = blockIdx.x * 8 + (threadIdx.x >> 5); // 0..511
  int lane = threadIdx.x & 31, half = lane >> 4, nl = lane & 15;
  int ntile = wid >> 2, dt = wid & 3;
  int row = ntile * 16 + nl;
  v8f c8 = {};
#pragma unroll
  for (int s = 0; s < 4; ++s) {
    v16h bf = *(const v16h*)(pooled + (size_t)row * 128 + s * 32 + 16 * half);
    v16h a  = *(const v16h*)(wAp + ((s * 4 + dt) * 32 + lane) * 16);
    c8 = wmma32(a, bf, c8);
  }
  int b = row >> 6, sp = row & 63;
  int d0 = dt * 16 + 8 * half;
  v8h o;
#pragma unroll
  for (int r = 0; r < 8; ++r) {
    float v = c8[r] + proj_b[d0 + r];
    ze_out[(size_t)b * 4096 + (size_t)(d0 + r) * 64 + sp] = v;  // z_e NCHW f32
    o[r] = (_Float16)v;
  }
  *(v8h*)(zf16 + (size_t)row * 64 + d0) = o;
}

// ---------------------------------------------------------------------------
// K4: VQ argmin over cnorm - 2*z.c via WMMA tiles + shfl_xor reduction.
// ---------------------------------------------------------------------------
__global__ __launch_bounds__(256) void k_vq(
    const _Float16* __restrict__ zf16, const _Float16* __restrict__ cbf16,
    const float* __restrict__ cnorm, int* __restrict__ idxws,
    float* __restrict__ idx_out)
{
  int wid = blockIdx.x * 8 + (threadIdx.x >> 5); // 0..127 = row tile
  int lane = threadIdx.x & 31, half = lane >> 4, nl = lane & 15;
  int row = wid * 16 + nl;
  const _Float16* zr = zf16 + (size_t)row * 64;
  v16h a0 = *(const v16h*)(zr + 16 * half);
  v16h a1 = *(const v16h*)(zr + 32 + 16 * half);

  float best[8]; int bidx[8];
#pragma unroll
  for (int r = 0; r < 8; ++r) { best[r] = 3.4e38f; bidx[r] = 0; }
  for (int ct = 0; ct < 32; ++ct) {
    int code = ct * 16 + nl;
    const _Float16* cr = cbf16 + (size_t)code * 64;
    v16h b0 = *(const v16h*)(cr + 16 * half);
    v16h b1 = *(const v16h*)(cr + 32 + 16 * half);
    v8f c8 = {};
    c8 = wmma32(a0, b0, c8);
    c8 = wmma32(a1, b1, c8);
    float cn = cnorm[code];
#pragma unroll
    for (int r = 0; r < 8; ++r) {
      float d = cn - 2.f * c8[r];
      if (d < best[r]) { best[r] = d; bidx[r] = code; }
    }
  }
  // min-reduce across the 16 lanes of each half (rows differ between halves)
#pragma unroll
  for (int off = 8; off > 0; off >>= 1) {
#pragma unroll
    for (int r = 0; r < 8; ++r) {
      float ov = __shfl_xor(best[r], off);
      int   oi = __shfl_xor(bidx[r], off);
      if (ov < best[r] || (ov == best[r] && oi < bidx[r])) { best[r] = ov; bidx[r] = oi; }
    }
  }
  if (nl == 0) {
#pragma unroll
    for (int r = 0; r < 8; ++r) {
      int rr = wid * 16 + r + 8 * half;
      idxws[rr] = bidx[r];
      idx_out[rr] = (float)bidx[r];
    }
  }
}

// ---------------------------------------------------------------------------
// K5: vq_loss = 1.25 * mean((codebook[idx] - z_e)^2), deterministic reduction
// ---------------------------------------------------------------------------
__global__ __launch_bounds__(256) void k_loss1(
    const float* __restrict__ codebook, const float* __restrict__ ze_out,
    const int* __restrict__ idxws, float* __restrict__ partial)
{
  __shared__ float sm[256];
  int n = blockIdx.x * 256 + threadIdx.x;
  int b = n >> 6, sp = n & 63;
  const float* cr = codebook + (size_t)idxws[n] * 64;
  float acc = 0.f;
  for (int d = 0; d < 64; ++d) {
    float diff = cr[d] - ze_out[(size_t)b * 4096 + (size_t)d * 64 + sp];
    acc += diff * diff;
  }
  sm[threadIdx.x] = acc;
  __syncthreads();
  for (int s = 128; s > 0; s >>= 1) {
    if (threadIdx.x < (unsigned)s) sm[threadIdx.x] += sm[threadIdx.x + s];
    __syncthreads();
  }
  if (threadIdx.x == 0) partial[blockIdx.x] = sm[0];
}

__global__ void k_loss2(const float* __restrict__ partial, float* __restrict__ loss_out)
{
  if (threadIdx.x == 0) {
    float s = 0.f;
    for (int i = 0; i < 8; ++i) s += partial[i];
    *loss_out = s * (1.25f / (2048.f * 64.f));
  }
}

// ---------------------------------------------------------------------------
// K6: dproj 1x1 64->128 on z_q = codebook[idx], ReLU -> padded ufp f16
// (32,10,10,128), interior blocks (gh+1, gw+1); border pre-zeroed.
// ---------------------------------------------------------------------------
__global__ __launch_bounds__(256) void k_dproj(
    const _Float16* __restrict__ cbf16, const int* __restrict__ idxws,
    const float* __restrict__ dproj_b, const _Float16* __restrict__ wAdp,
    _Float16* __restrict__ ufp)
{
  int wid = blockIdx.x * 8 + (threadIdx.x >> 5); // 0..1023
  int lane = threadIdx.x & 31, half = lane >> 4, nl = lane & 15;
  int ntile = wid >> 3, ht = wid & 7;
  int row = ntile * 16 + nl;
  const _Float16* cr = cbf16 + (size_t)idxws[row] * 64;
  v8f c8 = {};
#pragma unroll
  for (int s = 0; s < 2; ++s) {
    v16h bf = *(const v16h*)(cr + s * 32 + 16 * half);
    v16h a  = *(const v16h*)(wAdp + ((s * 8 + ht) * 32 + lane) * 16);
    c8 = wmma32(a, bf, c8);
  }
  int h0 = ht * 16 + 8 * half;
  v8h o;
#pragma unroll
  for (int r = 0; r < 8; ++r)
    o[r] = (_Float16)fmaxf(c8[r] + dproj_b[h0 + r], 0.f);
  int b = row >> 6, s = row & 63;
  int gh = s >> 3, gw = s & 7;
  *(v8h*)(ufp + ((size_t)b * 100 + (gh + 1) * 10 + gw + 1) * 128 + h0) = o;
}

// ---------------------------------------------------------------------------
// K7: dconv1 3x3 p1 128->128 + ReLU on the *virtual* 24x24 map (x3 upsample
// is conv-equivalent to x16 on piecewise-constant blocks). Dual pixel-tile
// per wave, fully unrolled, padded input -> no bounds checks.
// Out: tf NHWC f16 (32, 576, 128).
// ---------------------------------------------------------------------------
__global__ __launch_bounds__(256) void k_dconv1(
    const _Float16* __restrict__ ufp, const float* __restrict__ dconv1_b,
    const _Float16* __restrict__ wAd1r, _Float16* __restrict__ tf)
{
  int wid = blockIdx.x * 8 + (threadIdx.x >> 5);   // 4608 waves
  int lane = threadIdx.x & 31, half = lane >> 4, nl = lane & 15;
  int b = wid / 144;                     // 144 waves per batch (18 dual-tiles x 8)
  int rem = wid - b * 144;
  int vt = rem >> 3, oct = rem & 7;
  int vpx0 = vt * 32 + nl, vpx1 = vpx0 + 16;       // virtual pixels in 24x24
  int vr0 = vpx0 / 24, vc0 = vpx0 - vr0 * 24;
  int vr1 = vpx1 / 24, vc1 = vpx1 - vr1 * 24;
  // padded-block indices for each tap offset: (v-1+k+3)/3, pre-divided
  int pr0[3], pc0[3], pr1[3], pc1[3];
#pragma unroll
  for (int k = 0; k < 3; ++k) {
    pr0[k] = (vr0 + 2 + k) / 3; pc0[k] = (vc0 + 2 + k) / 3;
    pr1[k] = (vr1 + 2 + k) / 3; pc1[k] = (vc1 + 2 + k) / 3;
  }
  const _Float16* ub = ufp + (size_t)b * 100 * 128 + 16 * half;
  const _Float16* wab = wAd1r + (size_t)oct * 36 * 512 + lane * 16;
  v8f c0 = {}, c1 = {};
#pragma unroll
  for (int ki = 0; ki < 3; ++ki) {
#pragma unroll
    for (int kj = 0; kj < 3; ++kj) {
      int tap = ki * 3 + kj;
      const _Float16* p0 = ub + (pr0[ki] * 10 + pc0[kj]) * 128;
      const _Float16* p1 = ub + (pr1[ki] * 10 + pc1[kj]) * 128;
#pragma unroll
      for (int cs = 0; cs < 4; ++cs) {
        v16h a  = *(const v16h*)(wab + (tap * 4 + cs) * 512);
        v16h b0 = *(const v16h*)(p0 + cs * 32);
        v16h b1 = *(const v16h*)(p1 + cs * 32);
        c0 = wmma32(a, b0, c0);
        c1 = wmma32(a, b1, c1);
      }
    }
  }
  int oc0 = oct * 16 + 8 * half;
  v8h o0, o1;
#pragma unroll
  for (int r = 0; r < 8; ++r) {
    float bb = dconv1_b[oc0 + r];
    o0[r] = (_Float16)fmaxf(c0[r] + bb, 0.f);
    o1[r] = (_Float16)fmaxf(c1[r] + bb, 0.f);
  }
  *(v8h*)(tf + ((size_t)b * 576 + vpx0) * 128 + oc0) = o0;
  *(v8h*)(tf + ((size_t)b * 576 + vpx1) * 128 + oc0) = o1;
}

// ---------------------------------------------------------------------------
// K8: dconv2 1x1 128->3 + broadcast virtual pixels to full (32,3,128,128)
// ---------------------------------------------------------------------------
__global__ __launch_bounds__(256) void k_out(
    const _Float16* __restrict__ tf, const float* __restrict__ dconv2_w,
    const float* __restrict__ dconv2_b, float* __restrict__ xhat)
{
  int t = blockIdx.x * 256 + threadIdx.x; // 0..524287 = (b, i, j)
  int b = t >> 14, p = t & 16383;
  int i = p >> 7, j = p & 127;
  int pi = i & 15, pj = j & 15;
  int ri = (pi == 0) ? 0 : ((pi == 15) ? 2 : 1);
  int rj = (pj == 0) ? 0 : ((pj == 15) ? 2 : 1);
  int vr = 3 * (i >> 4) + ri, vc = 3 * (j >> 4) + rj;
  const _Float16* base = tf + (size_t)(b * 576 + vr * 24 + vc) * 128;
  float a0 = 0.f, a1 = 0.f, a2 = 0.f;
  for (int c = 0; c < 128; c += 8) {
    v8h t8 = *(const v8h*)(base + c);
#pragma unroll
    for (int u = 0; u < 8; ++u) {
      float v = (float)t8[u];
      a0 += v * dconv2_w[c + u];
      a1 += v * dconv2_w[128 + c + u];
      a2 += v * dconv2_w[256 + c + u];
    }
  }
  size_t ob = (size_t)b * 49152 + p;
  xhat[ob]         = a0 + dconv2_b[0];
  xhat[ob + 16384] = a1 + dconv2_b[1];
  xhat[ob + 32768] = a2 + dconv2_b[2];
}

// ---------------------------------------------------------------------------
extern "C" void kernel_launch(void* const* d_in, const int* in_sizes, int n_in,
                              void* d_out, int out_size, void* d_ws, size_t ws_size,
                              hipStream_t stream)
{
  (void)in_sizes; (void)n_in; (void)out_size; (void)ws_size;
  const float* x        = (const float*)d_in[0];
  const float* enc_w1   = (const float*)d_in[1];
  const float* enc_b1   = (const float*)d_in[2];
  const float* enc_w2   = (const float*)d_in[3];
  const float* enc_b2   = (const float*)d_in[4];
  const float* proj_w   = (const float*)d_in[5];
  const float* proj_b   = (const float*)d_in[6];
  const float* codebook = (const float*)d_in[7];
  const float* dproj_w  = (const float*)d_in[8];
  const float* dproj_b  = (const float*)d_in[9];
  const float* dconv1_w = (const float*)d_in[10];
  const float* dconv1_b = (const float*)d_in[11];
  const float* dconv2_w = (const float*)d_in[12];
  const float* dconv2_b = (const float*)d_in[13];

  // d_out: x_hat (1572864) | idx as float (2048) | vq_loss (1) | z_e (131072)
  float* out      = (float*)d_out;
  float* xhat     = out;
  float* idx_out  = out + 1572864;
  float* loss_out = out + 1574912;
  float* ze_out   = out + 1574913;

  char* ws = (char*)d_ws;
  _Float16* wA1    = (_Float16*)(ws + 0);        //   8 KB
  _Float16* wA2r   = (_Float16*)(ws + 8192);     // 288 KB
  _Float16* wAd1r  = (_Float16*)(ws + 303104);   // 288 KB
  _Float16* wAp    = (_Float16*)(ws + 598016);   //  16 KB
  _Float16* wAdp   = (_Float16*)(ws + 614400);   //  16 KB
  _Float16* cbf16  = (_Float16*)(ws + 630784);   //  64 KB
  float*    cnorm  = (float*)   (ws + 696320);   //   2 KB
  int*      idxws  = (int*)     (ws + 698368);   //   8 KB
  float*    partial= (float*)   (ws + 706560);   //  256 B
  _Float16* pooled = (_Float16*)(ws + 706816);   // 512 KB
  _Float16* zf16   = (_Float16*)(ws + 1231104);  // 256 KB
  _Float16* ufp    = (_Float16*)(ws + 1493248);  // 800 KB (32,10,10,128) padded
  _Float16* h1p    = (_Float16*)(ws + 2312448);  // 34.03 MB (32,66,66,128) padded
  _Float16* h2     = (_Float16*)(ws + 37996800); //  8.39 MB (32,32,32,128)
  _Float16* tf     = (_Float16*)(ws + 46385408); //  4.72 MB (32,24,24,128)

  k_repack<<<79, 256, 0, stream>>>(enc_w1, enc_w2, proj_w, dproj_w, dconv1_w,
                                   codebook, wA1, wA2r, wAd1r, wAp, wAdp, cbf16, cnorm);
  k_zero  <<<200, 256, 0, stream>>>((v8h*)ufp, 51200);        // zero padded u
  k_border<<<520, 256, 0, stream>>>(h1p);                     // zero h1 border
  k_enc1  <<<8192, 256, 0, stream>>>(x, enc_b1, wA1, h1p);
  k_enc2  <<<1024, 256, 0, stream>>>(h1p, enc_b2, wA2r, h2);
  k_pool  <<<2048, 128, 0, stream>>>(h2, pooled);
  k_proj  <<<64,   256, 0, stream>>>(pooled, proj_b, wAp, ze_out, zf16);
  k_vq    <<<16,   256, 0, stream>>>(zf16, cbf16, cnorm, idxws, idx_out);
  k_loss1 <<<8,    256, 0, stream>>>(codebook, ze_out, idxws, partial);
  k_loss2 <<<1,    32,  0, stream>>>(partial, loss_out);
  k_dproj <<<128,  256, 0, stream>>>(cbf16, idxws, dproj_b, wAdp, ufp);
  k_dconv1<<<576,  256, 0, stream>>>(ufp, dconv1_b, wAd1r, tf);
  k_out   <<<2048, 256, 0, stream>>>(tf, dconv2_w, dconv2_b, xhat);
}
